// PerspTransDetector_max_cvcs_48687749268006
// MI455X (gfx1250) — compile-verified, matrix operand order checked by reading, not run on hardware
//
#include <hip/hip_runtime.h>
#include <hip/hip_bf16.h>

#define EPSF 1e-5f

typedef __attribute__((ext_vector_type(16))) _Float16 v16h;
typedef __attribute__((ext_vector_type(8)))  _Float16 v8h;
typedef __attribute__((ext_vector_type(8)))  float    v8f;

// K index inside a 32-wide chunk for 16-bit A/B WMMA fragments (ISA 7.12.2):
// lanes 0-15:  vgpr0..3 -> K 0..7,  vgpr4..7 -> K 16..23
// lanes 16-31: vgpr0..3 -> K 8..15, vgpr4..7 -> K 24..31
static __device__ __forceinline__ int klocal(int slot, int half, int kg) {
  return ((slot < 4) ? 0 : 16) + kg * 8 + ((slot & 3) << 1) + half;
}
static __device__ __forceinline__ v16h cat8(v8h lo, v8h hi) {
  return __builtin_shufflevector(lo, hi, 0, 1, 2, 3, 4, 5, 6, 7,
                                 8, 9, 10, 11, 12, 13, 14, 15);
}
static __device__ __forceinline__ v8f wmma16(v16h a, v16h b, v8f c) {
  return __builtin_amdgcn_wmma_f32_16x16x32_f16(false, a, false, b, (short)0, c,
                                                false, false);
}

// ---------------------------------------------------------------------------
// Weight repack into WMMA B-fragment order: Wp[((tn*kChunks+kc)*32+lane)*16+h]
// K ordering for conv2d: k = (kh*KS+kw)*Cin + ci  (tap-major, ci minor)
// ---------------------------------------------------------------------------
__global__ void repack2d_k(const float* __restrict__ wgt, _Float16* __restrict__ Wp,
                           int Cout, int Cin, int KK, int Ktot, int kChunks, int nTiles) {
  long idx = (long)blockIdx.x * blockDim.x + threadIdx.x;
  long tot = (long)nTiles * kChunks * 512;
  if (idx >= tot) return;
  int h = idx & 15;
  int lane = (idx >> 4) & 31;
  long t = idx >> 9;
  int kc = t % kChunks;
  int tn = t / kChunks;
  int kg = lane >> 4, slot = h >> 1, half = h & 1;
  int k = kc * 32 + klocal(slot, half, kg);
  int n = tn * 16 + (lane & 15);
  float v = 0.f;
  if (n < Cout && k < Ktot) {
    int tap = k / Cin, ci = k - tap * Cin;
    v = wgt[((long)n * Cin + ci) * KK + tap];
  }
  Wp[idx] = (_Float16)v;
}

// conv4d weights are already K-major: wgt[k*Cout + co], k = tap*CIN + ci
__global__ void repack4d_k(const float* __restrict__ wgt, _Float16* __restrict__ Wp,
                           int Cout, int Ktot, int kChunks) {
  long idx = (long)blockIdx.x * blockDim.x + threadIdx.x;
  long tot = (long)kChunks * 512;
  if (idx >= tot) return;
  int h = idx & 15;
  int lane = (idx >> 4) & 31;
  int kc = idx >> 9;
  int kg = lane >> 4, slot = h >> 1, half = h & 1;
  int k = kc * 32 + klocal(slot, half, kg);
  int co = lane & 15;
  float v = 0.f;
  if (co < Cout && k < Ktot) v = wgt[(long)k * Cout + co];
  Wp[idx] = (_Float16)v;
}

// ---------------------------------------------------------------------------
// Fast implicit-GEMM conv: NHWC f16, Cin%32==0, HW%16==0.
// One wave computes a 16(spatial) x 64(cout) tile: the predicated A fragment
// (2 x b128) is reused by 4 WMMAs against 4 pre-packed B fragments.
// K loop is tap x cin-chunk nested (no division, bounds hoisted per tap).
// ---------------------------------------------------------------------------
template <int KS>
__global__ void conv2d_fast(const _Float16* __restrict__ in, const _Float16* __restrict__ Wp,
                            const float* __restrict__ bias, _Float16* __restrict__ out,
                            int N, int Cin, int H, int W, int Cout, int dil, int relu) {
  const int lane = threadIdx.x & 31;
  const int wave = threadIdx.x >> 5;
  const int HW = H * W;
  const int mTiles = HW >> 4;
  const int nTiles = (Cout + 15) >> 4;
  const int nGroups = (nTiles + 3) >> 2;
  int tile = blockIdx.x * (blockDim.x >> 5) + wave;
  if (tile >= N * mTiles * nGroups) return;
  int n = tile / (mTiles * nGroups);
  int r = tile - n * (mTiles * nGroups);
  int tm = r / nGroups, tg = r - tm * nGroups;
  int tn0 = tg * 4;
  const int nt = (nTiles - tn0 < 4) ? (nTiles - tn0) : 4;  // wave-uniform
  const int row = lane & 15, kg = lane >> 4;
  int m = tm * 16 + row;
  int oh = m / W, ow = m - oh * W;
  const int cpt = Cin >> 5;  // cin chunks per tap
  const int kChunks = KS * KS * cpt;
  const long tstride = (long)kChunks * 512;
  const _Float16* inN = in + (long)n * HW * Cin;
  const _Float16* wp0 = Wp + (long)tn0 * tstride + (long)lane * 16;

  v8f acc0 = {}, acc1 = {}, acc2 = {}, acc3 = {};
  int kc = 0;
  for (int tap = 0; tap < KS * KS; ++tap) {
    int kh = tap / KS, kw = tap - kh * KS;  // KS compile-time: cheap
    int ih = oh + (kh - KS / 2) * dil;
    int iw = ow + (kw - KS / 2) * dil;
    bool ok = (ih >= 0 && ih < H && iw >= 0 && iw < W);
    const _Float16* ap = inN + ((long)ih * W + iw) * Cin + kg * 8;
    for (int c = 0; c < cpt; ++c, ++kc) {
      v8h alo = {}, ahi = {};
      if (ok) {
        alo = *(const v8h*)(ap + c * 32);
        ahi = *(const v8h*)(ap + c * 32 + 16);
      }
      v16h a = cat8(alo, ahi);
      const _Float16* wk = wp0 + (long)kc * 512;
      if (kc + 1 < kChunks) __builtin_prefetch(wk + 512, 0, 1);
      acc0 = wmma16(a, *(const v16h*)(wk), acc0);
      if (nt > 1) acc1 = wmma16(a, *(const v16h*)(wk + tstride), acc1);
      if (nt > 2) acc2 = wmma16(a, *(const v16h*)(wk + 2 * tstride), acc2);
      if (nt > 3) acc3 = wmma16(a, *(const v16h*)(wk + 3 * tstride), acc3);
    }
  }
  v8f accs[4] = {acc0, acc1, acc2, acc3};
  for (int t = 0; t < nt; ++t) {
    int cc = (tn0 + t) * 16 + row;
    if (cc >= Cout) break;
    float bs = bias[cc];
#pragma unroll
    for (int rv = 0; rv < 8; ++rv) {
      int mm = tm * 16 + rv + kg * 8;
      float v = accs[t][rv] + bs;
      if (relu) v = fmaxf(v, 0.f);
      out[((long)n * HW + mm) * Cout + cc] = (_Float16)v;
    }
  }
}

// Gather conv for layer 0 (Cin=3, Ktot=27, 1 chunk), compile-time Cin/KS,
// same 4-wide N-blocking (Cout=64 -> one wave covers all couts of 16 pixels).
template <int CIN, int KS>
__global__ void conv2d_gather(const _Float16* __restrict__ in, const _Float16* __restrict__ Wp,
                              const float* __restrict__ bias, _Float16* __restrict__ out,
                              int N, int H, int W, int Cout, int dil, int relu) {
  const int lane = threadIdx.x & 31;
  const int wave = threadIdx.x >> 5;
  const int HW = H * W;
  const int mTiles = HW >> 4;
  const int nTiles = (Cout + 15) >> 4;
  const int nGroups = (nTiles + 3) >> 2;
  int tile = blockIdx.x * (blockDim.x >> 5) + wave;
  if (tile >= N * mTiles * nGroups) return;
  int n = tile / (mTiles * nGroups);
  int r = tile - n * (mTiles * nGroups);
  int tm = r / nGroups, tg = r - tm * nGroups;
  int tn0 = tg * 4;
  const int nt = (nTiles - tn0 < 4) ? (nTiles - tn0) : 4;
  const int row = lane & 15, kg = lane >> 4;
  int m = tm * 16 + row;
  int oh = m / W, ow = m - oh * W;
  const int Ktot = KS * KS * CIN;
  const int kChunks = (Ktot + 31) >> 5;
  const long tstride = (long)kChunks * 512;
  const _Float16* inN = in + (long)n * HW * CIN;
  const _Float16* wp0 = Wp + (long)tn0 * tstride + (long)lane * 16;

  v8f acc0 = {}, acc1 = {}, acc2 = {}, acc3 = {};
  for (int kc = 0; kc < kChunks; ++kc) {
    v16h a;
#pragma unroll
    for (int slot = 0; slot < 8; ++slot) {
#pragma unroll
      for (int half = 0; half < 2; ++half) {
        int k = kc * 32 + klocal(slot, half, kg);
        _Float16 av = (_Float16)0.f;
        if (k < Ktot) {
          int tap = k / CIN, ci = k - tap * CIN;  // CIN compile-time
          int kh = tap / KS, kw = tap - kh * KS;
          int ih = oh + (kh - KS / 2) * dil;
          int iw = ow + (kw - KS / 2) * dil;
          if (ih >= 0 && ih < H && iw >= 0 && iw < W)
            av = inN[((long)ih * W + iw) * CIN + ci];
        }
        a[slot * 2 + half] = av;
      }
    }
    const _Float16* wk = wp0 + (long)kc * 512;
    acc0 = wmma16(a, *(const v16h*)(wk), acc0);
    if (nt > 1) acc1 = wmma16(a, *(const v16h*)(wk + tstride), acc1);
    if (nt > 2) acc2 = wmma16(a, *(const v16h*)(wk + 2 * tstride), acc2);
    if (nt > 3) acc3 = wmma16(a, *(const v16h*)(wk + 3 * tstride), acc3);
  }
  v8f accs[4] = {acc0, acc1, acc2, acc3};
  for (int t = 0; t < nt; ++t) {
    int cc = (tn0 + t) * 16 + row;
    if (cc >= Cout) break;
    float bs = bias[cc];
#pragma unroll
    for (int rv = 0; rv < 8; ++rv) {
      int mm = tm * 16 + rv + kg * 8;
      float v = accs[t][rv] + bs;
      if (relu) v = fmaxf(v, 0.f);
      out[((long)n * HW + mm) * Cout + cc] = (_Float16)v;
    }
  }
}

// ---------------------------------------------------------------------------
// Correlation GEMM: C(784x784) = Braw(784x256) x Araw(256x784), f16 inputs.
// fB rows contiguous; fAT is the pre-transposed copy so B side is contiguous.
// ---------------------------------------------------------------------------
__global__ void corr_gemm_wmma(const _Float16* __restrict__ fB,
                               const _Float16* __restrict__ fAT,
                               float* __restrict__ corr) {
  const int lane = threadIdx.x & 31;
  const int wave = threadIdx.x >> 5;
  int tile = blockIdx.x * (blockDim.x >> 5) + wave;
  if (tile >= 49 * 49) return;
  int tm = tile / 49, tn = tile - tm * 49;
  int row = lane & 15, kg = lane >> 4;
  const _Float16* ap = fB + (long)(tm * 16 + row) * 256;
  const _Float16* bp = fAT + (long)(tn * 16 + row) * 256;
  v8f acc = {};
#pragma unroll
  for (int kc = 0; kc < 8; ++kc) {
    int k0 = kc * 32;
    v16h a = cat8(*(const v8h*)(ap + k0 + kg * 8), *(const v8h*)(ap + k0 + 16 + kg * 8));
    v16h b = cat8(*(const v8h*)(bp + k0 + kg * 8), *(const v8h*)(bp + k0 + 16 + kg * 8));
    acc = wmma16(a, b, acc);
  }
  int jj = tn * 16 + row;
#pragma unroll
  for (int rv = 0; rv < 8; ++rv) {
    int mm = tm * 16 + rv + kg * 8;
    corr[mm * 784 + jj] = acc[rv];
  }
}

// ---------------------------------------------------------------------------
// 4D conv (SAME, 3^4 taps), channels-last f16, gather A, packed B fragments.
// ---------------------------------------------------------------------------
template <int CIN>
__global__ void conv4d_wmma(const _Float16* __restrict__ x, const _Float16* __restrict__ Wp,
                            const float* __restrict__ bias, _Float16* __restrict__ out,
                            int N, int Cout) {
  const int S = 28, S4 = 614656;
  const int lane = threadIdx.x & 31;
  const int wave = threadIdx.x >> 5;
  int tile = blockIdx.x * (blockDim.x >> 5) + wave;
  const int mTiles = S4 / 16;
  if (tile >= N * mTiles) return;
  int n = tile / mTiles;
  int tm = tile - n * mTiles;
  int row = lane & 15, kg = lane >> 4;
  int m = tm * 16 + row;
  int e = m % S;  int t = m / S;
  int d = t % S;  t /= S;
  int bb = t % S; int aa = t / S;
  const int Ktot = 81 * CIN;
  const int kChunks = (Ktot + 31) >> 5;
  const _Float16* xn = x + (long)n * S4 * CIN;

  v8f acc = {};
  for (int kc = 0; kc < kChunks; ++kc) {
    v16h a;
#pragma unroll
    for (int slot = 0; slot < 8; ++slot) {
#pragma unroll
      for (int half = 0; half < 2; ++half) {
        int k = kc * 32 + klocal(slot, half, kg);
        _Float16 av = (_Float16)0.f;
        if (k < Ktot) {
          int ci = k % CIN; int tt = k / CIN;  // CIN compile-time
          int te = tt % 3; tt /= 3;
          int td = tt % 3; tt /= 3;
          int tb = tt % 3; int ta = tt / 3;
          int ia = aa + ta - 1, ib = bb + tb - 1, id = d + td - 1, ie = e + te - 1;
          if ((unsigned)ia < 28u && (unsigned)ib < 28u && (unsigned)id < 28u &&
              (unsigned)ie < 28u)
            av = xn[((long)(((ia * S + ib) * S + id) * S + ie)) * CIN + ci];
        }
        a[slot * 2 + half] = av;
      }
    }
    v16h b = *(const v16h*)(Wp + ((long)kc * 32 + lane) * 16);
    acc = wmma16(a, b, acc);
  }
  int cc = lane & 15;
  if (cc < Cout) {
    float bs = bias[cc];
#pragma unroll
    for (int rv = 0; rv < 8; ++rv) {
      long mm = tm * 16 + rv + kg * 8;
      out[((long)n * S4 + mm) * Cout + cc] = (_Float16)fmaxf(acc[rv] + bs, 0.f);
    }
  }
}

// ------------------------- data-layout / small kernels ---------------------
__global__ void img2nhwc_f16_k(const float* __restrict__ in, _Float16* __restrict__ out) {
  long idx = (long)blockIdx.x * blockDim.x + threadIdx.x;  // 3*224*224*3
  if (idx >= 3L * 224 * 224 * 3) return;
  int c = idx % 3; long t = idx / 3;
  int w = t % 224; t /= 224;
  int h = t % 224; int n = t / 224;
  out[idx] = (_Float16)in[(((long)n * 3 + c) * 224 + h) * 224 + w];
}

__global__ void maxpool_nhwc_k(const _Float16* __restrict__ in, _Float16* __restrict__ out,
                               int N, int H, int W, int C) {
  int Ho = H >> 1, Wo = W >> 1;
  long idx = (long)blockIdx.x * blockDim.x + threadIdx.x;
  long tot = (long)N * Ho * Wo * C;
  if (idx >= tot) return;
  int c = idx % C; long t = idx / C;
  int ow = t % Wo; t /= Wo;
  int oh = t % Ho; int n = t / Ho;
  const _Float16* p = in + (((long)n * H + oh * 2) * W + ow * 2) * C + c;
  float v0 = (float)p[0], v1 = (float)p[C];
  float v2 = (float)p[(long)W * C], v3 = (float)p[(long)W * C + C];
  out[idx] = (_Float16)fmaxf(fmaxf(v0, v1), fmaxf(v2, v3));
}

// L2-normalize pooled NHWC (3,28,28,256) per pixel; emit NCHW f16 buffer
// (NCHW order reproduces the reference's raw-reshape semantics)
__global__ void l2norm_f16_k(const _Float16* __restrict__ in, _Float16* __restrict__ out,
                             int N, int C, int HW) {
  long idx = (long)blockIdx.x * blockDim.x + threadIdx.x;
  if (idx >= (long)N * HW) return;
  int s = idx % HW; int n = idx / HW;
  const _Float16* p = in + ((long)n * HW + s) * C;
  float ss = 0.f;
  for (int c = 0; c < C; ++c) { float v = (float)p[c]; ss += v * v; }
  float inv = 1.f / (sqrtf(ss) + EPSF);
  for (int c = 0; c < C; ++c)
    out[((long)n * C + c) * HW + s] = (_Float16)((float)p[c] * inv);
}

// nfT[n][j][k] = nf[n][k*784 + j]  (so corr-GEMM B side is contiguous)
__global__ void transp_k(const _Float16* __restrict__ nf, _Float16* __restrict__ nfT) {
  long idx = (long)blockIdx.x * blockDim.x + threadIdx.x;  // 3*256*784
  if (idx >= 3L * 200704) return;
  long rem = idx % 200704; int n = idx / 200704;
  int k = rem / 784, j = rem % 784;
  nfT[(long)n * 200704 + (long)j * 256 + k] = nf[idx];
}

template <typename T>
__global__ void rowmax_k(const T* __restrict__ y, float* __restrict__ rmax,
                         int nmat, int rows, int cols) {
  int idx = blockIdx.x * blockDim.x + threadIdx.x;
  if (idx >= nmat * rows) return;
  const T* p = y + (long)idx * cols;
  float m = -INFINITY;
  for (int j = 0; j < cols; ++j) m = fmaxf(m, (float)p[j]);
  rmax[idx] = m;
}

template <typename T>
__global__ void colmax_k(const T* __restrict__ y, float* __restrict__ cmax,
                         int nmat, int rows, int cols) {
  int idx = blockIdx.x * blockDim.x + threadIdx.x;
  if (idx >= nmat * cols) return;
  int j = idx % cols; int n = idx / cols;
  const T* p = y + (long)n * rows * cols + j;
  float m = -INFINITY;
  for (int i = 0; i < rows; ++i) m = fmaxf(m, (float)p[(long)i * cols]);
  cmax[idx] = m;
}

// out = corr^3 / ((rowmax+eps)*(colmax+eps)) -> f16 (x4d input)
__global__ void corr_prod_k(const float* __restrict__ c, const float* __restrict__ rmax,
                            const float* __restrict__ cmax, _Float16* __restrict__ out) {
  long idx = (long)blockIdx.x * blockDim.x + threadIdx.x;
  if (idx >= 784L * 784L) return;
  int i = idx / 784, j = idx % 784;
  float v = c[idx];
  out[idx] = (_Float16)(v * v * v / ((rmax[i] + EPSF) * (cmax[j] + EPSF)));
}

// wij = max over softmax(row of cA) = 1 / sum exp(v - vmax)
__global__ void wij_k(const _Float16* __restrict__ y, const float* __restrict__ rmax,
                      float* __restrict__ wcat) {
  int idx = blockIdx.x * blockDim.x + threadIdx.x;  // (n,i) over 3*784
  if (idx >= 3 * 784) return;
  const _Float16* row = y + (long)idx * 784;
  float dnm = rmax[idx] + EPSF;
  float m = rmax[idx] / dnm;
  float s = 0.f;
  for (int j = 0; j < 784; ++j) s += __expf((float)row[j] / dnm - m);
  wcat[idx * 2 + 0] = 1.f / s;
}

// per (n,i,h2): softmax denominators of cB over w2 (last 28)
__global__ void cbden_k(const _Float16* __restrict__ y, const float* __restrict__ cmax,
                        float* __restrict__ marr, float* __restrict__ sarr) {
  int idx = blockIdx.x * blockDim.x + threadIdx.x;  // 3*784*28
  if (idx >= 3 * 784 * 28) return;
  int h2 = idx % 28; int t = idx / 28;
  int i = t % 784;   int n = t / 784;
  const _Float16* row = y + ((long)(n * 784 + i)) * 784 + h2 * 28;
  const float* cm = cmax + n * 784 + h2 * 28;
  float v[28], m = -INFINITY;
  for (int w2 = 0; w2 < 28; ++w2) {
    v[w2] = (float)row[w2] / (cm[w2] + EPSF);
    m = fmaxf(m, v[w2]);
  }
  float s = 0.f;
  for (int w2 = 0; w2 < 28; ++w2) s += __expf(v[w2] - m);
  marr[idx] = m; sarr[idx] = s;
}

// wji = argmax over i of softmaxed cB at column j=(h2,w2)
__global__ void wji_k(const _Float16* __restrict__ y, const float* __restrict__ cmax,
                      const float* __restrict__ marr, const float* __restrict__ sarr,
                      float* __restrict__ wcat) {
  int idx = blockIdx.x * blockDim.x + threadIdx.x;  // (n,j) over 3*784
  if (idx >= 3 * 784) return;
  int n = idx / 784; int j = idx % 784; int h2 = j / 28;
  float cm = cmax[idx] + EPSF;
  float best = -INFINITY; int bi = 0;
  for (int i = 0; i < 784; ++i) {
    int ni = n * 784 + i;
    float val = __expf((float)y[(long)ni * 784 + j] / cm - marr[ni * 28 + h2]) /
                sarr[ni * 28 + h2];
    if (val > best) { best = val; bi = i; }
  }
  wcat[idx * 2 + 1] = (float)bi;
}

__global__ void wmap_k(const float* __restrict__ wcat, float* __restrict__ wmap) {
  int idx = blockIdx.x * blockDim.x + threadIdx.x;  // 6*784
  if (idx >= 6 * 784) return;
  int hw = idx % 784; int ch = idx / 784;
  int p = ch >> 1, c = ch & 1;
  int i1 = (p == 2) ? 1 : 0;
  int i2 = (p == 0) ? 1 : 2;
  float a = wcat[(i1 * 784 + hw) * 2 + c];
  float b = wcat[(i2 * 784 + hw) * 2 + c];
  wmap[idx] = 1.f / (1.f + a + b);
}

__global__ void final_sum_k(const _Float16* __restrict__ imgres,
                            const float* __restrict__ wmap, float* __restrict__ out) {
  __shared__ float red[256];
  float acc = 0.f;
  for (int idx = threadIdx.x; idx < 3 * 6 * 784; idx += 256) {
    int hw = idx % 784; int t = idx / 784;
    int ch = t % 6;     int n = t / 6;
    acc += ((float)imgres[n * 784 + hw] * 0.01f) * wmap[ch * 784 + hw];
  }
  red[threadIdx.x] = acc;
  __syncthreads();
  for (int s = 128; s > 0; s >>= 1) {
    if ((int)threadIdx.x < s) red[threadIdx.x] += red[threadIdx.x + s];
    __syncthreads();
  }
  if (threadIdx.x == 0) out[0] = red[0];
}

// --------------------------- host orchestration ----------------------------
static inline int cdivi(long a, long b) { return (int)((a + b - 1) / b); }

static void conv2dL(hipStream_t s, const _Float16* in, const float* wgt, const float* bias,
                    _Float16* out, _Float16* wbuf, int N, int Cin, int H, int W, int Cout,
                    int dil, int relu, int ks) {
  int KK = ks * ks, Ktot = KK * Cin;
  int kChunks = (Ktot + 31) / 32, nTiles = (Cout + 15) / 16;
  int nGroups = (nTiles + 3) / 4;
  long relems = (long)nTiles * kChunks * 512;
  repack2d_k<<<cdivi(relems, 256), 256, 0, s>>>(wgt, wbuf, Cout, Cin, KK, Ktot, kChunks,
                                                nTiles);
  long tiles = (long)N * (H * W / 16) * nGroups;
  int blocks = cdivi(tiles, 4);
  if ((Cin & 31) == 0) {
    if (ks == 3)
      conv2d_fast<3><<<blocks, 128, 0, s>>>(in, wbuf, bias, out, N, Cin, H, W, Cout, dil, relu);
    else
      conv2d_fast<1><<<blocks, 128, 0, s>>>(in, wbuf, bias, out, N, Cin, H, W, Cout, dil, relu);
  } else {
    conv2d_gather<3, 3><<<blocks, 128, 0, s>>>(in, wbuf, bias, out, N, H, W, Cout, dil, relu);
  }
}

extern "C" void kernel_launch(void* const* d_in, const int* in_sizes, int n_in,
                              void* d_out, int out_size, void* d_ws, size_t ws_size,
                              hipStream_t stream) {
  (void)in_sizes; (void)n_in; (void)out_size; (void)ws_size;
  const float* imgs = (const float*)d_in[0];
  auto W_ = [&](int base, int i) { return (const float*)d_in[base + 2 * i]; };
  auto B_ = [&](int base, int i) { return (const float*)d_in[base + 2 * i + 1]; };
  const int BASE = 1, CLS = 15, CORR = 35, C4D = 49;

  // ---- workspace carve-out (256B aligned regions) ----
  char* wsp = (char*)d_ws;
  auto alloc = [&](size_t bytes) {
    char* p = wsp;
    wsp += (bytes + 255) & ~(size_t)255;
    return p;
  };
  _Float16* P0h   = (_Float16*)alloc(18439680ULL * 2);  // (3,784,784,10) max
  _Float16* P1h   = (_Float16*)alloc(18439680ULL * 2);
  float*    corrf = (float*)   alloc(614656ULL * 4);
  _Float16* img16 = (_Float16*)alloc(451584ULL * 2);
  _Float16* wbuf  = (_Float16*)alloc(2359296ULL * 2);   // max repacked layer
  _Float16* nf    = (_Float16*)alloc(602112ULL * 2);
  _Float16* nfT   = (_Float16*)alloc(602112ULL * 2);
  float*    rmaxb = (float*)   alloc(2352 * 4);
  float*    cmaxb = (float*)   alloc(2352 * 4);
  float*    marr  = (float*)   alloc(65856 * 4);
  float*    sarr  = (float*)   alloc(65856 * 4);
  _Float16* ires  = (_Float16*)alloc(2352 * 2);

  float* out = (float*)d_out;
  float* wmapO = out + 1;
  float* wcatO = out + 1 + 4704;
  const int EB = 256, N = 3;

  img2nhwc_f16_k<<<cdivi(451584L, EB), EB, 0, stream>>>(imgs, img16);

  // ---- base encoder (NHWC f16) ----
  conv2dL(stream, img16, W_(BASE,0), B_(BASE,0), P0h, wbuf, N,   3, 224, 224,  64, 1, 1, 3);
  conv2dL(stream, P0h,   W_(BASE,1), B_(BASE,1), P1h, wbuf, N,  64, 224, 224,  64, 1, 1, 3);
  maxpool_nhwc_k<<<cdivi(3L*112*112*64, EB), EB, 0, stream>>>(P1h, P0h, N, 224, 224, 64);
  conv2dL(stream, P0h,   W_(BASE,2), B_(BASE,2), P1h, wbuf, N,  64, 112, 112, 128, 1, 1, 3);
  conv2dL(stream, P1h,   W_(BASE,3), B_(BASE,3), P0h, wbuf, N, 128, 112, 112, 128, 1, 1, 3);
  maxpool_nhwc_k<<<cdivi(3L*56*56*128, EB), EB, 0, stream>>>(P0h, P1h, N, 112, 112, 128);
  conv2dL(stream, P1h,   W_(BASE,4), B_(BASE,4), P0h, wbuf, N, 128,  56,  56, 256, 1, 1, 3);
  conv2dL(stream, P0h,   W_(BASE,5), B_(BASE,5), P1h, wbuf, N, 256,  56,  56, 256, 1, 1, 3);
  conv2dL(stream, P1h,   W_(BASE,6), B_(BASE,6), P0h, wbuf, N, 256,  56,  56, 256, 1, 1, 3);

  // ---- classifier ----
  {
    const int cin[10]  = {256,512,512,512,512,512,512,256,128,64};
    const int cout[10] = {512,512,512,512,512,512,256,128, 64, 1};
    const int dil[10]  = {1,1,1,2,2,2,2,2,2,1};
    _Float16* src = P0h; _Float16* dst = P1h;
    for (int i = 0; i < 10; ++i) {
      conv2dL(stream, src, W_(CLS,i), B_(CLS,i), dst, wbuf, N, cin[i], 56, 56, cout[i],
              dil[i], i < 9 ? 1 : 0, (i == 9) ? 1 : 3);
      _Float16* t = src; src = dst; dst = t;
    }
    maxpool_nhwc_k<<<cdivi(3L*28*28, EB), EB, 0, stream>>>(src, ires, N, 56, 56, 1);
  }

  // ---- corr encoder ----
  conv2dL(stream, img16, W_(CORR,0), B_(CORR,0), P0h, wbuf, N,   3, 224, 224,  64, 1, 1, 3);
  conv2dL(stream, P0h,   W_(CORR,1), B_(CORR,1), P1h, wbuf, N,  64, 224, 224,  64, 1, 1, 3);
  maxpool_nhwc_k<<<cdivi(3L*112*112*64, EB), EB, 0, stream>>>(P1h, P0h, N, 224, 224, 64);
  conv2dL(stream, P0h,   W_(CORR,2), B_(CORR,2), P1h, wbuf, N,  64, 112, 112, 128, 1, 1, 3);
  conv2dL(stream, P1h,   W_(CORR,3), B_(CORR,3), P0h, wbuf, N, 128, 112, 112, 128, 1, 1, 3);
  maxpool_nhwc_k<<<cdivi(3L*56*56*128, EB), EB, 0, stream>>>(P0h, P1h, N, 112, 112, 128);
  conv2dL(stream, P1h,   W_(CORR,4), B_(CORR,4), P0h, wbuf, N, 128,  56,  56, 256, 1, 1, 3);
  conv2dL(stream, P0h,   W_(CORR,5), B_(CORR,5), P1h, wbuf, N, 256,  56,  56, 256, 1, 1, 3);
  conv2dL(stream, P1h,   W_(CORR,6), B_(CORR,6), P0h, wbuf, N, 256,  56,  56, 256, 1, 1, 3);

  // pool to 28x28, L2-normalize into NCHW-f16, build transposed copy
  maxpool_nhwc_k<<<cdivi(3L*28*28*256, EB), EB, 0, stream>>>(P0h, P1h, N, 56, 56, 256);
  l2norm_f16_k<<<cdivi(3L*784, EB), EB, 0, stream>>>(P1h, nf, N, 256, 784);
  transp_k<<<cdivi(3L*200704, EB), EB, 0, stream>>>(nf, nfT);

  // ---- 3 pairwise correlations -> x4d input in P0h (3,784,784) f16 ----
  {
    const int pb[3] = {0, 0, 1}, pa[3] = {1, 2, 2};
    for (int p = 0; p < 3; ++p) {
      corr_gemm_wmma<<<cdivi(49L * 49L, 4), 128, 0, stream>>>(
          nf + (long)pb[p] * 200704, nfT + (long)pa[p] * 200704, corrf);
      rowmax_k<float><<<cdivi(784, EB), EB, 0, stream>>>(corrf, rmaxb, 1, 784, 784);
      colmax_k<float><<<cdivi(784, EB), EB, 0, stream>>>(corrf, cmaxb, 1, 784, 784);
      corr_prod_k<<<cdivi(784L * 784L, EB), EB, 0, stream>>>(corrf, rmaxb, cmaxb,
                                                             P0h + (long)p * 614656L);
    }
  }

  // ---- conv4d stack: 1->10->10->1, relu each ----
  {
    int blocks = cdivi(3L * 38416L, 4);
    repack4d_k<<<cdivi(3L * 512, 256), 256, 0, stream>>>(W_(C4D,0), wbuf, 10, 81, 3);
    conv4d_wmma<1> <<<blocks, 128, 0, stream>>>(P0h, wbuf, B_(C4D,0), P1h, N, 10);
    repack4d_k<<<cdivi(26L * 512, 256), 256, 0, stream>>>(W_(C4D,1), wbuf, 10, 810, 26);
    conv4d_wmma<10><<<blocks, 128, 0, stream>>>(P1h, wbuf, B_(C4D,1), P0h, N, 10);
    repack4d_k<<<cdivi(26L * 512, 256), 256, 0, stream>>>(W_(C4D,2), wbuf, 1, 810, 26);
    conv4d_wmma<10><<<blocks, 128, 0, stream>>>(P0h, wbuf, B_(C4D,2), P1h, N, 1);
  }
  const _Float16* y = P1h;  // (3,784,784) f16

  // ---- final reductions ----
  rowmax_k<_Float16><<<cdivi(3 * 784, EB), EB, 0, stream>>>(y, rmaxb, 3, 784, 784);
  colmax_k<_Float16><<<cdivi(3 * 784, EB), EB, 0, stream>>>(y, cmaxb, 3, 784, 784);
  wij_k<<<cdivi(3 * 784, EB), EB, 0, stream>>>(y, rmaxb, wcatO);
  cbden_k<<<cdivi(3 * 784 * 28, EB), EB, 0, stream>>>(y, cmaxb, marr, sarr);
  wji_k<<<cdivi(3 * 784, EB), EB, 0, stream>>>(y, cmaxb, marr, sarr, wcatO);
  wmap_k<<<cdivi(6 * 784, EB), EB, 0, stream>>>(wcatO, wmapO);
  final_sum_k<<<1, 256, 0, stream>>>(ires, wmapO, out);
}
